// CausalSelfAttention_46411416600586
// MI455X (gfx1250) — matrix operand order repaired and checked by reading, not compiled, not run on hardware
//
#include <hip/hip_runtime.h>

#define T_SEQ 2048
#define NB    4
#define NH    16
#define DH    64
#define CDIM  1024

typedef __attribute__((ext_vector_type(8)))  float  v8f;
typedef __attribute__((ext_vector_type(16))) __bf16 v16bf;

__device__ __forceinline__ unsigned short f2bf(float f) {
  unsigned int u = __float_as_uint(f);
  u += 0x7FFFu + ((u >> 16) & 1u);          // round-to-nearest-even
  return (unsigned short)(u >> 16);
}

// A-fragment (16x32 bf16): lane reads row m, k = {2*ko..2*ko+15}
__device__ __forceinline__ v16bf load_frag_a(const unsigned short* row, int ko) {
  union { v16bf v; uint4 u[2]; } t;
  t.u[0] = *(const uint4*)(row + 2 * ko);
  t.u[1] = *(const uint4*)(row + 2 * ko + 8);
  return t.v;
}
// 16 contiguous bf16 (B-fragment slice: K-dim contiguous in memory)
__device__ __forceinline__ v16bf load_frag_c16(const unsigned short* p) {
  union { v16bf v; uint4 u[2]; } t;
  t.u[0] = *(const uint4*)(p);
  t.u[1] = *(const uint4*)(p + 8);
  return t.v;
}

// Async global->LDS copy of 16 bytes (CDNA5 GLOBAL_LOAD_ASYNC_TO_LDS_B128,
// GV mode, tracked by ASYNCcnt).  lds_off is the wave-relative LDS byte
// address = low 32 bits of the generic pointer (ISA 10.2 aperture layout).
__device__ __forceinline__ void async_copy_b128(const void* gsrc, void* lds_dst) {
  unsigned lds_off = (unsigned)(size_t)lds_dst;
  unsigned long long ga = (unsigned long long)(size_t)gsrc;
  asm volatile("global_load_async_to_lds_b128 %0, %1, off"
               :: "v"(lds_off), "v"(ga) : "memory");
}
__device__ __forceinline__ void wait_async0() {
  asm volatile("s_wait_asynccnt 0x0" ::: "memory");
}

__global__ void cast_f32_to_bf16(const float* __restrict__ in,
                                 unsigned short* __restrict__ out, int n) {
  int i = blockIdx.x * blockDim.x + threadIdx.x;
  int stride = gridDim.x * blockDim.x;
  for (; i < n; i += stride) out[i] = f2bf(in[i]);
}

// ---------------------------------------------------------------------------
// bf16 GEMM: C[M,N] = A[M,K] * B[K,N].  Block tile 128(M) x 64(N), 8 waves,
// each wave: one 16-row strip x 4 N-tiles of 16.
//   A tile [128x32] staged via async global->LDS (coalesced 16B/lane bursts),
//   B tile staged in LDS pre-swizzled into a contiguous-k B-fragment layout matching A
//   (lanes 0-15: K=0..15, lanes 16-31: K=16..31, VGPR v holds K=2v,2v+1).
// mode 0: scatter epilogue -> Q[B,H,T,Dh], K[B,H,T,Dh], Vt[B,H,Dh,T] (bf16)
// mode 1: plain fp32 store -> outF[M,N]
// ---------------------------------------------------------------------------
__global__ void gemm_bf16_kernel(const unsigned short* __restrict__ A,
                                 const unsigned short* __restrict__ Bm,
                                 int M, int N, int K, int mode,
                                 unsigned short* __restrict__ outQ,
                                 unsigned short* __restrict__ outK,
                                 unsigned short* __restrict__ outVt,
                                 float* __restrict__ outF) {
  __shared__ __attribute__((aligned(32))) unsigned short lds_b[4 * 32 * 16]; // 4KB frag-layout
  __shared__ __attribute__((aligned(32))) unsigned short lds_a[128 * 32];    // 8KB row-major

  const int tid  = threadIdx.x;
  const int wave = tid >> 5;
  const int lane = tid & 31;
  const int ln16 = lane & 15;
  const int hi   = lane >> 4;           // 0 or 1 (half-wave)
  const int blockM = blockIdx.y * 128;
  const int blockN = blockIdx.x * 64;
  const int mrow   = blockM + wave * 16;
  const int ko_a   = hi * 8;

  v8f acc[4];
#pragma unroll
  for (int t = 0; t < 4; ++t) acc[t] = (v8f){0.f,0.f,0.f,0.f,0.f,0.f,0.f,0.f};

  for (int kb = 0; kb < K; kb += 32) {
    // -------- async A tile copy: [128 rows x 64B] global -> LDS --------
#pragma unroll
    for (int it = 0; it < 2; ++it) {
      int q = tid + it * 256;           // chunk id 0..511
      int r = q >> 2;                   // tile row 0..127
      int c = q & 3;                    // 16B chunk in row
      async_copy_b128(A + (size_t)(blockM + r) * K + kb + c * 8,
                      &lds_a[r * 32 + c * 8]);
    }

    // -------- cooperative B tile load + swizzle into fragment layout --------
    {
      int row = tid >> 3;               // 0..31 (local K)
      int c8  = (tid & 7) * 8;          // 0..56 (local N)
      const unsigned short* src = Bm + (size_t)(kb + row) * N + blockN + c8;
      uint4 raw = *(const uint4*)src;   // 8 bf16
      const unsigned short* e = (const unsigned short*)&raw;
      int laneoff = row & 16;           // K>=16 -> lanes 16..31
      int vg   = (row & 15) >> 1;
      int half = row & 1;
#pragma unroll
      for (int j = 0; j < 8; ++j) {
        int nl = c8 + j;
        int ntile = nl >> 4;
        int blane = (nl & 15) + laneoff;
        lds_b[(((ntile << 5) + blane) << 4) + (vg << 1) + half] = e[j];
      }
      if (kb + 32 < K)                  // prefetch next K-step's B tile
        __builtin_prefetch((const void*)(src + 32 * (size_t)N), 0, 1);
    }

    wait_async0();                      // this wave's A-tile chunks landed
    __syncthreads();                    // all waves' A+B staging visible

    // -------- fragments from LDS --------
    const unsigned short* arow = &lds_a[(wave * 16 + ln16) * 32];
    v16bf af = load_frag_a(arow, ko_a);

#pragma unroll
    for (int t = 0; t < 4; ++t) {
      v16bf bf = load_frag_c16(&lds_b[((t << 5) + lane) << 4]);
      acc[t] = __builtin_amdgcn_wmma_f32_16x16x32_bf16(
          false, af, false, bf, (short)0, acc[t], false, false);
    }
    __syncthreads();
  }

  // -------- epilogue --------
  if (mode == 0) {
#pragma unroll
    for (int t = 0; t < 4; ++t) {
#pragma unroll
      for (int v = 0; v < 8; ++v) {
        int m = mrow + v + hi * 8;
        int n = blockN + t * 16 + ln16;
        unsigned short hv = f2bf(acc[t][v]);
        int b   = m >> 11;              // T_SEQ = 2048
        int tt  = m & (T_SEQ - 1);
        int seg = n >> 10;              // 0=q 1=k 2=v
        int c   = n & (CDIM - 1);
        int h   = c >> 6;
        int dh  = c & (DH - 1);
        size_t qk = ((((size_t)b * NH + h) * T_SEQ) + tt) * DH + dh;
        if (seg == 0)      outQ[qk] = hv;
        else if (seg == 1) outK[qk] = hv;
        else outVt[(((size_t)b * NH + h) * DH + dh) * T_SEQ + tt] = hv;
      }
    }
  } else {
#pragma unroll
    for (int t = 0; t < 4; ++t) {
#pragma unroll
      for (int v = 0; v < 8; ++v) {
        int m = mrow + v + hi * 8;
        int n = blockN + t * 16 + ln16;
        outF[(size_t)m * N + n] = acc[t][v];
      }
    }
  }
}

// ---------------------------------------------------------------------------
// Flash attention: grid (T/128, B*H), 8 waves/block, each wave owns a
// 16-query tile.  Streams 32-key blocks: S = QK^T (2 WMMA per 16-key N-tile,
// K-dim = Dh split 32+32), online softmax with shfl_xor row reductions,
// P -> LDS -> A-fragment, O += P V (4 WMMA, V pre-transposed [Dh,T]).
// Writes bf16 attention output in [B,T,C] layout for the final projection.
// ---------------------------------------------------------------------------
__global__ void flash_attn_kernel(const unsigned short* __restrict__ Q,
                                  const unsigned short* __restrict__ Km,
                                  const unsigned short* __restrict__ Vt,
                                  unsigned short* __restrict__ Out) {
  __shared__ __attribute__((aligned(32))) unsigned short p_lds[8][16 * 32];

  const int tid  = threadIdx.x;
  const int wave = tid >> 5;
  const int lane = tid & 31;
  const int ln16 = lane & 15;
  const int hi   = lane >> 4;
  const int bh   = blockIdx.y;               // 0..63
  const int b    = bh >> 4, h = bh & 15;
  const int qb   = blockIdx.x * 128 + wave * 16;

  const size_t headQK = (size_t)bh * T_SEQ * DH;   // [T,Dh] base
  const size_t headV  = (size_t)bh * DH * T_SEQ;   // [Dh,T] base
  const int ko_a = hi * 8;
  const int ko_b = hi * 16;
  const float scale = 0.125f;                // 1/sqrt(64)

  // Q A-fragments (Dh 0..31 and 32..63), loaded once
  const unsigned short* qrow = Q + headQK + (size_t)(qb + ln16) * DH;
  v16bf qf0 = load_frag_a(qrow, ko_a);
  v16bf qf1 = load_frag_a(qrow + 32, ko_a);

  float mrow[8], lrow[8];
  v8f o[4];
#pragma unroll
  for (int v = 0; v < 8; ++v) { mrow[v] = -1e30f; lrow[v] = 0.f; }
#pragma unroll
  for (int t = 0; t < 4; ++t) o[t] = (v8f){0.f,0.f,0.f,0.f,0.f,0.f,0.f,0.f};

  for (int kb = 0; kb < qb + 16; kb += 32) {
    // ---- S = Q K^T for two 16-key tiles ----
    v8f s0 = (v8f){0.f,0.f,0.f,0.f,0.f,0.f,0.f,0.f};
    v8f s1 = s0;
    {
      const unsigned short* krow = Km + headQK + (size_t)(kb + ln16) * DH;
      v16bf k0 = load_frag_c16(krow + ko_b);
      v16bf k1 = load_frag_c16(krow + 32 + ko_b);
      s0 = __builtin_amdgcn_wmma_f32_16x16x32_bf16(false, qf0, false, k0, (short)0, s0, false, false);
      s0 = __builtin_amdgcn_wmma_f32_16x16x32_bf16(false, qf1, false, k1, (short)0, s0, false, false);
    }
    {
      const unsigned short* krow = Km + headQK + (size_t)(kb + 16 + ln16) * DH;
      v16bf k0 = load_frag_c16(krow + ko_b);
      v16bf k1 = load_frag_c16(krow + 32 + ko_b);
      s1 = __builtin_amdgcn_wmma_f32_16x16x32_bf16(false, qf0, false, k0, (short)0, s1, false, false);
      s1 = __builtin_amdgcn_wmma_f32_16x16x32_bf16(false, qf1, false, k1, (short)0, s1, false, false);
    }

    // ---- scale + causal mask + per-row block max ----
    float bm[8];
#pragma unroll
    for (int v = 0; v < 8; ++v) {
      int qi  = qb + v + hi * 8;
      int k0i = kb + ln16;
      int k1i = kb + 16 + ln16;
      float a0 = s0[v] * scale; if (k0i > qi) a0 = -1e30f;
      float a1 = s1[v] * scale; if (k1i > qi) a1 = -1e30f;
      s0[v] = a0; s1[v] = a1;
      bm[v] = fmaxf(a0, a1);
    }
#pragma unroll
    for (int v = 0; v < 8; ++v) {       // row max across the 16-lane half
      float x = bm[v];
      x = fmaxf(x, __shfl_xor(x, 1));
      x = fmaxf(x, __shfl_xor(x, 2));
      x = fmaxf(x, __shfl_xor(x, 4));
      x = fmaxf(x, __shfl_xor(x, 8));
      bm[v] = x;
    }

    // ---- online softmax update ----
    float alpha[8];
#pragma unroll
    for (int v = 0; v < 8; ++v) {
      float mn = fmaxf(mrow[v], bm[v]);
      alpha[v] = __expf(mrow[v] - mn);
      mrow[v] = mn;
      float p0 = __expf(s0[v] - mn);
      float p1 = __expf(s1[v] - mn);
      s0[v] = p0; s1[v] = p1;
      float ps = p0 + p1;
      ps += __shfl_xor(ps, 1);
      ps += __shfl_xor(ps, 2);
      ps += __shfl_xor(ps, 4);
      ps += __shfl_xor(ps, 8);
      lrow[v] = lrow[v] * alpha[v] + ps;
    }
#pragma unroll
    for (int t = 0; t < 4; ++t)
#pragma unroll
      for (int v = 0; v < 8; ++v) o[t][v] *= alpha[v];

    // ---- P: C-layout -> A-fragment via wave-private LDS ----
    unsigned short* pw = p_lds[wave];
#pragma unroll
    for (int v = 0; v < 8; ++v) {
      int m = v + hi * 8;
      pw[m * 32 + ln16]      = f2bf(s0[v]);
      pw[m * 32 + 16 + ln16] = f2bf(s1[v]);
    }
    asm volatile("s_wait_dscnt 0x0" ::: "memory");
    v16bf pf = load_frag_a(pw + ln16 * 32, ko_a);

    // ---- O += P V  (V transposed: B-fragment is contiguous along keys) ----
#pragma unroll
    for (int t = 0; t < 4; ++t) {
      const unsigned short* vrow =
          Vt + headV + (size_t)(t * 16 + ln16) * T_SEQ + kb + ko_b;
      v16bf bv = load_frag_c16(vrow);
      o[t] = __builtin_amdgcn_wmma_f32_16x16x32_bf16(
          false, pf, false, bv, (short)0, o[t], false, false);
    }
  }

  // ---- normalize and write [B,T,C] bf16 ----
#pragma unroll
  for (int t = 0; t < 4; ++t) {
#pragma unroll
    for (int v = 0; v < 8; ++v) {
      int m = v + hi * 8;
      int q = qb + m;
      float val = o[t][v] / lrow[v];
      Out[((size_t)b * T_SEQ + q) * CDIM + h * DH + t * 16 + ln16] = f2bf(val);
    }
  }
}

// ---------------------------------------------------------------------------
extern "C" void kernel_launch(void* const* d_in, const int* in_sizes, int n_in,
                              void* d_out, int out_size, void* d_ws, size_t ws_size,
                              hipStream_t stream) {
  (void)in_sizes; (void)n_in; (void)out_size; (void)ws_size;
  const float* x    = (const float*)d_in[0];   // [4,2048,1024]
  const float* Wqkv = (const float*)d_in[1];   // [1024,3072]
  const float* Wout = (const float*)d_in[2];   // [1024,1024]
  float* out = (float*)d_out;

  char* ws = (char*)d_ws;
  const size_t N_X    = (size_t)NB * T_SEQ * CDIM;        // 8,388,608
  const size_t N_WQKV = (size_t)CDIM * 3 * CDIM;          // 3,145,728
  const size_t N_WOUT = (size_t)CDIM * CDIM;              // 1,048,576

  unsigned short* xb    = (unsigned short*)(ws);                    // also attn-out
  unsigned short* wqkvb = (unsigned short*)(ws + 2 * N_X);
  unsigned short* woutb = (unsigned short*)((char*)wqkvb + 2 * N_WQKV);
  unsigned short* Qb    = (unsigned short*)((char*)woutb + 2 * N_WOUT);
  unsigned short* Kb    = Qb + N_X;
  unsigned short* Vtb   = Kb + N_X;

  // 1) casts fp32 -> bf16
  cast_f32_to_bf16<<<8192, 256, 0, stream>>>(x,    xb,    (int)N_X);
  cast_f32_to_bf16<<<4096, 256, 0, stream>>>(Wqkv, wqkvb, (int)N_WQKV);
  cast_f32_to_bf16<<<2048, 256, 0, stream>>>(Wout, woutb, (int)N_WOUT);

  // 2) QKV projection, scatter into Q/K (row-major) and V (transposed)
  gemm_bf16_kernel<<<dim3(48, 64), 256, 0, stream>>>(
      xb, wqkvb, NB * T_SEQ, 3 * CDIM, CDIM, 0, Qb, Kb, Vtb, nullptr);

  // 3) flash attention -> bf16 [B,T,C] (reuses xb region; xb is dead now)
  flash_attn_kernel<<<dim3(T_SEQ / 128, NB * NH), 256, 0, stream>>>(
      Qb, Kb, Vtb, xb);

  // 4) output projection -> fp32
  gemm_bf16_kernel<<<dim3(16, 64), 256, 0, stream>>>(
      xb, woutb, NB * T_SEQ, CDIM, CDIM, 1, nullptr, nullptr, nullptr, out);
}
